// RGCNSkipConnection_42949673547
// MI455X (gfx1250) — compile-verified, matrix operand order checked by reading, not run on hardware
//
#include <hip/hip_runtime.h>

// ---------------- problem constants (match reference) ----------------
#define N_NODES   50000
#define N_EDGES   800000
#define N_REL     6
#define N_LAYERS  8
#define DDIM      256
#define NUM_GRAPHS 512

typedef __bf16 bf16;
typedef __attribute__((ext_vector_type(16))) __bf16 v16bf;
typedef __attribute__((ext_vector_type(8)))  __bf16 v8bf;
typedef __attribute__((ext_vector_type(8)))  float  v8f;

typedef __attribute__((ext_vector_type(4))) unsigned int u32x4;
typedef __attribute__((ext_vector_type(8))) int          i32x8;
typedef __attribute__((ext_vector_type(4))) int          i32x4;

union V16U { v16bf v; v8bf h[2]; };

#if __has_builtin(__builtin_amdgcn_tensor_load_to_lds) && __has_builtin(__builtin_amdgcn_s_wait_tensorcnt)
#define USE_TDM 1
#endif

// ---------------------------------------------------------------------
// WMMA GEMM: C[M,N] (f32) = A[M,K] (bf16, row-major) * Bt[N,K]^T (bf16)
// epilogue: + bias[col] (optional), optional ReLU.
// Block = 256 threads = 8 waves. Block stages Bt slice [blockIdx.y*64 .. +63][0..K)
// into LDS (TDM async DMA when available), then wave w computes rows
// [(blockIdx.x*8+w)*16 .. +15] x 64 cols as 4 x 16x16 f32 accumulators.
// Requires: M % 16 == 0, N % 64 == 0, K % 32 == 0.
// Dynamic LDS: 64*K*2 bytes.
// ---------------------------------------------------------------------
__global__ __launch_bounds__(256)
void wmma_gemm_bf16(const bf16* __restrict__ A, const bf16* __restrict__ Bt,
                    float* __restrict__ C, const float* __restrict__ bias,
                    int M, int N, int K, int relu)
{
    extern __shared__ bf16 sB[];                   // [64][K], LDS offset 0

    const int wave = threadIdx.x >> 5;
    const int lane = threadIdx.x & 31;
    const int l    = lane & 15;
    const int half = lane >> 4;
    const int tile_n = blockIdx.y;                 // 64 cols

    // ---- stage Bt[tile_n*64 .. +63][0..K) into LDS ------------------
#ifdef USE_TDM
    if (wave == 0) {
        // Tensor DMA descriptor (CDNA5 ISA):
        // 2D tile: tile_dim0 = K/2 dwords (row of K bf16), tile_dim1 = 64 rows.
        unsigned long long gaddr =
            (unsigned long long)(const void*)(Bt + (size_t)tile_n * 64 * K);
        unsigned int dw = (unsigned int)(K >> 1);  // dwords per row

        u32x4 g0;
        g0[0] = 1u;                                // count=1 (valid), no gather
        g0[1] = 0u;                                // lds_addr = 0 (dynamic-LDS base)
        g0[2] = (unsigned int)(gaddr & 0xFFFFFFFFull);          // global_addr lo
        g0[3] = (unsigned int)((gaddr >> 32) & 0x1FFFFFFull)    // global_addr[56:32]
                | 0x80000000u;                                  // type = 2 (image)

        i32x8 g1;
        g1[0] = 0x20000;                           // data_size = 2 (4-byte units)
        g1[1] = (int)((dw & 0xFFFFu) << 16);       // tensor_dim0 lo16 (atomic_bar=0)
        g1[2] = (int)((dw >> 16) | (64u << 16));   // tensor_dim0 hi16 | tensor_dim1 lo16
        g1[3] = (int)((dw & 0xFFFFu) << 16);       // tensor_dim1 hi16=0 | tile_dim0
        g1[4] = 64;                                // tile_dim1 = 64, tile_dim2 = 0
        g1[5] = (int)dw;                           // tensor_dim0_stride lo32
        g1[6] = 0;                                 // stride0 hi16 | stride1 lo16
        g1[7] = 0;                                 // stride1 hi32

        i32x4 g2 = {0, 0, 0, 0};
        i32x4 g3 = {0, 0, 0, 0};
        i32x8 g4 = {0, 0, 0, 0, 0, 0, 0, 0};       // 6-arg toolchain: extra group, zero
        __builtin_amdgcn_tensor_load_to_lds(g0, g1, g2, g3, g4, 0);
        __builtin_amdgcn_s_wait_tensorcnt(0);
    }
    __syncthreads();
#else
    {
        const v8bf* __restrict__ gsrc = (const v8bf*)(Bt + (size_t)tile_n * 64 * K);
        v8bf* lds = (v8bf*)sB;
        const int nchunks = (64 * K) >> 3;
        for (int idx = threadIdx.x; idx < nchunks; idx += 256) lds[idx] = gsrc[idx];
        __syncthreads();
    }
#endif

    const int tile_m = blockIdx.x * 8 + wave;      // 16 rows
    if (tile_m * 16 >= M) return;                  // wave-uniform exit (EXEC all-1)

    const bf16* __restrict__ arow = A + (size_t)(tile_m * 16 + l) * K;

    v8f acc[4];
    acc[0] = v8f{}; acc[1] = v8f{}; acc[2] = v8f{}; acc[3] = v8f{};

    for (int k0 = 0; k0 < K; k0 += 32) {
        // A fragment: 16x32 bf16 per ISA layout — two contiguous 16B chunks/lane
        V16U a;
        a.h[0] = *(const v8bf*)(arow + k0 + 8 * half);
        a.h[1] = *(const v8bf*)(arow + k0 + 16 + 8 * half);
        if (k0 + 32 < K) __builtin_prefetch(arow + k0 + 32, 0, 1);

#pragma unroll
        for (int t = 0; t < 4; ++t) {
            const bf16* __restrict__ bcol = sB + (size_t)(t * 16 + l) * K;
            V16U b;
            b.h[0] = *(const v8bf*)(bcol + k0 + 8 * half);
            b.h[1] = *(const v8bf*)(bcol + k0 + 16 + 8 * half);
            acc[t] = __builtin_amdgcn_wmma_f32_16x16x32_bf16(
                false, a.v, false, b.v, (short)0, acc[t], false, false);
        }
    }

    // C/D layout: VGPR i of lane -> row = 8*half + i, col = lane&15
    const int r0 = tile_m * 16 + 8 * half;
#pragma unroll
    for (int t = 0; t < 4; ++t) {
        const int col = tile_n * 64 + t * 16 + l;
        const float bb = bias ? bias[col] : 0.0f;
#pragma unroll
        for (int i = 0; i < 8; ++i) {
            float v = acc[t][i] + bb;
            if (relu) v = fmaxf(v, 0.0f);
            C[(size_t)(r0 + i) * N + col] = v;
        }
    }
}

// ---------------- encoder: h = x @ enc_W + enc_b ; also bf16 copy ----
__global__ __launch_bounds__(256)
void encoder_kernel(const float* __restrict__ x, const float* __restrict__ W,
                    const float* __restrict__ b, float* __restrict__ h,
                    bf16* __restrict__ hb)
{
    int idx = blockIdx.x * 256 + threadIdx.x;      // n*D + d
    if (idx >= N_NODES * DDIM) return;
    int n = idx >> 8, d = idx & 255;
    float acc = b[d];
#pragma unroll
    for (int j = 0; j < 13; ++j) acc += x[n * 13 + j] * W[j * DDIM + d];
    h[idx] = acc;
    hb[idx] = (bf16)acc;
}

// ------------- batched transpose + cast: W[K,N] f32 -> Wt[N,K] bf16 --
__global__ __launch_bounds__(256)
void transpose_cast_kernel(const float* __restrict__ W, bf16* __restrict__ Wt,
                           int K, int N)
{
    size_t mat = blockIdx.z;
    const float* w = W + mat * (size_t)K * N;
    bf16* o = Wt + mat * (size_t)K * N;
    int idx = blockIdx.x * 256 + threadIdx.x;
    if (idx >= K * N) return;
    int k = idx / N, n = idx - k * N;
    o[(size_t)n * K + k] = (bf16)w[idx];
}

// ------------- elementwise f32 -> bf16 -------------------------------
__global__ __launch_bounds__(256)
void cast_bf16_kernel(const float* __restrict__ in, bf16* __restrict__ out, long n)
{
    long idx = (long)blockIdx.x * 256 + threadIdx.x;
    if (idx < n) out[idx] = (bf16)in[idx];
}

__global__ __launch_bounds__(256)
void zero_kernel(float* __restrict__ p, long n)
{
    long idx = (long)blockIdx.x * 256 + threadIdx.x;
    if (idx < n) p[idx] = 0.0f;
}

// ------------- edge scatter for one relation -------------------------
// agg[dst[e]] += xw[src[e]]  for edges with edge_type == rel.
// 64 threads / edge, float4 per thread.
__global__ __launch_bounds__(256)
void scatter_rel_kernel(const float* __restrict__ xw, const int* __restrict__ src,
                        const int* __restrict__ dst, const int* __restrict__ etype,
                        int rel, float* __restrict__ agg)
{
    int e = blockIdx.x * 4 + (threadIdx.x >> 6);
    if (e >= N_EDGES) return;
    if (etype[e] != rel) return;
    int t = threadIdx.x & 63;
    float4 v = ((const float4*)(xw + (size_t)src[e] * DDIM))[t];
    float* d = agg + (size_t)dst[e] * DDIM + t * 4;
    atomicAdd(d + 0, v.x);
    atomicAdd(d + 1, v.y);
    atomicAdd(d + 2, v.z);
    atomicAdd(d + 3, v.w);
}

// ------------- layer epilogue: h = normalize(prelu(agg + h)) ---------
__global__ __launch_bounds__(256)
void layer_epilogue_kernel(const float* __restrict__ agg, float* __restrict__ h,
                           bf16* __restrict__ hb, const float* __restrict__ a_ptr)
{
    __shared__ float red[256];
    int row = blockIdx.x, t = threadIdx.x;
    size_t idx = (size_t)row * DDIM + t;
    float a = a_ptr[0];
    float v = agg[idx] + h[idx];
    v = v > 0.0f ? v : a * v;
    red[t] = v * v;
    __syncthreads();
#pragma unroll
    for (int s = 128; s > 0; s >>= 1) {
        if (t < s) red[t] += red[t + s];
        __syncthreads();
    }
    float inv = 1.0f / fmaxf(sqrtf(red[0]), 1e-12f);
    float o = v * inv;
    h[idx] = o;
    hb[idx] = (bf16)o;
}

// ------------- global add pool: pooled[batch[n]] += g[n] -------------
__global__ __launch_bounds__(256)
void pool_kernel(const float* __restrict__ g, const int* __restrict__ batch,
                 float* __restrict__ pooled)
{
    int idx = blockIdx.x * 256 + threadIdx.x;      // n*(D/4) + q
    if (idx >= N_NODES * (DDIM / 4)) return;
    int n = idx >> 6, q = idx & 63;
    int b = batch[n];
    float4 v = ((const float4*)(g + (size_t)n * DDIM))[q];
    float* d = pooled + (size_t)b * DDIM + q * 4;
    atomicAdd(d + 0, v.x);
    atomicAdd(d + 1, v.y);
    atomicAdd(d + 2, v.z);
    atomicAdd(d + 3, v.w);
}

// ------------- final: out = relu(z2 @ out_W + out_b), N=1 ------------
__global__ __launch_bounds__(256)
void out_kernel(const float* __restrict__ z, const float* __restrict__ W,
                const float* __restrict__ b, float* __restrict__ out)
{
    __shared__ float red[256];
    int row = blockIdx.x, t = threadIdx.x;
    const float* zr = z + (size_t)row * 512;
    red[t] = zr[t] * W[t] + zr[t + 256] * W[t + 256];
    __syncthreads();
#pragma unroll
    for (int s = 128; s > 0; s >>= 1) {
        if (t < s) red[t] += red[t + s];
        __syncthreads();
    }
    if (t == 0) out[row] = fmaxf(red[0] + b[0], 0.0f);
}

// =====================================================================
extern "C" void kernel_launch(void* const* d_in, const int* in_sizes, int n_in,
                              void* d_out, int out_size, void* d_ws, size_t ws_size,
                              hipStream_t stream)
{
    const float* x      = (const float*)d_in[0];
    const int*   eidx   = (const int*)d_in[1];
    const int*   src    = eidx;
    const int*   dst    = eidx + N_EDGES;
    const int*   etype  = (const int*)d_in[2];
    const int*   batch  = (const int*)d_in[3];
    const float* enc_W  = (const float*)d_in[4];
    const float* enc_b  = (const float*)d_in[5];
    const float* prelu_a= (const float*)d_in[6];
    const float* rel_W  = (const float*)d_in[7];
    const float* root_W = (const float*)d_in[8];
    const float* conv_b = (const float*)d_in[9];
    const float* gp_W1  = (const float*)d_in[10];
    const float* gp_b1  = (const float*)d_in[11];
    const float* gp_W2  = (const float*)d_in[12];
    const float* gp_b2  = (const float*)d_in[13];
    const float* fc_W1  = (const float*)d_in[14];
    const float* fc_b1  = (const float*)d_in[15];
    const float* fc_W2  = (const float*)d_in[16];
    const float* fc_b2  = (const float*)d_in[17];
    const float* out_W  = (const float*)d_in[18];
    const float* out_b  = (const float*)d_in[19];
    float* outp = (float*)d_out;

    const size_t ND = (size_t)N_NODES * DDIM;
    size_t off = 0;
    char* base = (char*)d_ws;
    auto wsa = [&](size_t bytes) -> char* {
        char* p = base + off;
        off += (bytes + 255) & ~(size_t)255;
        return p;
    };
    float* h       = (float*)wsa(ND * 4);
    bf16*  hb      = (bf16*) wsa(ND * 2);
    float* xw      = (float*)wsa(ND * 4);     // per-relation transform / t1 / reuse
    bf16*  xwb     = (bf16*) wsa(ND * 2);
    float* agg     = (float*)wsa(ND * 4);     // accumulator / g
    bf16*  wt_rel  = (bf16*) wsa((size_t)N_LAYERS * N_REL * DDIM * DDIM * 2);
    bf16*  wt_root = (bf16*) wsa((size_t)N_LAYERS * DDIM * DDIM * 2);
    bf16*  wt_gp1  = (bf16*) wsa((size_t)DDIM * DDIM * 2);
    bf16*  wt_gp2  = (bf16*) wsa((size_t)DDIM * DDIM * 2);
    bf16*  wt_fc1  = (bf16*) wsa((size_t)DDIM * 1024 * 2);
    bf16*  wt_fc2  = (bf16*) wsa((size_t)1024 * 512 * 2);
    float* pooled  = (float*)wsa((size_t)NUM_GRAPHS * DDIM * 4);
    bf16*  pooledb = (bf16*) wsa((size_t)NUM_GRAPHS * DDIM * 2);
    float* z1      = (float*)wsa((size_t)NUM_GRAPHS * 1024 * 4);
    bf16*  z1b     = (bf16*) wsa((size_t)NUM_GRAPHS * 1024 * 2);
    float* z2      = (float*)wsa((size_t)NUM_GRAPHS * 512 * 4);

    // ---- weight precompute: transpose + bf16 cast -------------------
    transpose_cast_kernel<<<dim3(256, 1, N_LAYERS * N_REL), 256, 0, stream>>>(rel_W,  wt_rel,  DDIM, DDIM);
    transpose_cast_kernel<<<dim3(256, 1, N_LAYERS),         256, 0, stream>>>(root_W, wt_root, DDIM, DDIM);
    transpose_cast_kernel<<<dim3(256, 1, 1),                256, 0, stream>>>(gp_W1,  wt_gp1,  DDIM, DDIM);
    transpose_cast_kernel<<<dim3(256, 1, 1),                256, 0, stream>>>(gp_W2,  wt_gp2,  DDIM, DDIM);
    transpose_cast_kernel<<<dim3(1024, 1, 1),               256, 0, stream>>>(fc_W1,  wt_fc1,  DDIM, 1024);
    transpose_cast_kernel<<<dim3(2048, 1, 1),               256, 0, stream>>>(fc_W2,  wt_fc2,  1024, 512);

    // ---- encoder ----------------------------------------------------
    encoder_kernel<<<N_NODES, 256, 0, stream>>>(x, enc_W, enc_b, h, hb);

    // GEMM grids: M=50000 -> 3125 M-tiles -> 391 blocks of 8 waves; N=256 -> 4
    const dim3 g_big(391, DDIM / 64);
    const int mat = DDIM * DDIM;
    const size_t lds256  = 64 * 256 * sizeof(bf16);    // 32 KB
    const size_t lds1024 = 64 * 1024 * sizeof(bf16);   // 128 KB (WGP has 320 KB)

    // ---- RGCN layers ------------------------------------------------
    for (int i = 0; i < N_LAYERS; ++i) {
        // agg = h @ root_W[i] + conv_b[i]
        wmma_gemm_bf16<<<g_big, 256, lds256, stream>>>(hb, wt_root + (size_t)i * mat,
                                                       agg, conv_b + (size_t)i * DDIM,
                                                       N_NODES, DDIM, DDIM, 0);
        for (int r = 0; r < N_REL; ++r) {
            // xw = h @ rel_W[i][r]
            wmma_gemm_bf16<<<g_big, 256, lds256, stream>>>(hb, wt_rel + (size_t)(i * N_REL + r) * mat,
                                                           xw, nullptr, N_NODES, DDIM, DDIM, 0);
            // agg[dst] += xw[src] for edges of relation r
            scatter_rel_kernel<<<N_EDGES / 4, 256, 0, stream>>>(xw, src, dst, etype, r, agg);
        }
        // h = normalize(prelu(agg + h))
        layer_epilogue_kernel<<<N_NODES, 256, 0, stream>>>(agg, h, hb, prelu_a);
    }

    // ---- graph head -------------------------------------------------
    // t1 = relu(h @ gp_W1 + gp_b1)
    wmma_gemm_bf16<<<g_big, 256, lds256, stream>>>(hb, wt_gp1, xw, gp_b1, N_NODES, DDIM, DDIM, 1);
    cast_bf16_kernel<<<(int)((ND + 255) / 256), 256, 0, stream>>>(xw, xwb, (long)ND);
    // g = t1 @ gp_W2 + gp_b2   (into agg)
    wmma_gemm_bf16<<<g_big, 256, lds256, stream>>>(xwb, wt_gp2, agg, gp_b2, N_NODES, DDIM, DDIM, 0);

    // pooled = segment_sum(g, batch)
    long pooledN = (long)NUM_GRAPHS * DDIM;
    zero_kernel<<<(int)((pooledN + 255) / 256), 256, 0, stream>>>(pooled, pooledN);
    pool_kernel<<<(N_NODES * (DDIM / 4) + 255) / 256, 256, 0, stream>>>(agg, batch, pooled);
    cast_bf16_kernel<<<(int)((pooledN + 255) / 256), 256, 0, stream>>>(pooled, pooledb, pooledN);

    // z1 = relu(pooled @ fc_W1 + fc_b1)   M=512 -> 32 tiles -> 4 blocks; N=1024 -> 16
    wmma_gemm_bf16<<<dim3(4, 16), 256, lds256, stream>>>(pooledb, wt_fc1, z1, fc_b1,
                                                         NUM_GRAPHS, 1024, DDIM, 1);
    long z1N = (long)NUM_GRAPHS * 1024;
    cast_bf16_kernel<<<(int)((z1N + 255) / 256), 256, 0, stream>>>(z1, z1b, z1N);

    // z2 = relu(z1 @ fc_W2 + fc_b2)   N=512 -> 8
    wmma_gemm_bf16<<<dim3(4, 8), 256, lds1024, stream>>>(z1b, wt_fc2, z2, fc_b2,
                                                         NUM_GRAPHS, 512, 1024, 1);

    // out = relu(z2 @ out_W + out_b)
    out_kernel<<<NUM_GRAPHS, 256, 0, stream>>>(z2, out_W, out_b, outp);
}
